// Attention_78812649881818
// MI455X (gfx1250) — matrix operand order repaired and checked by reading, not compiled, not run on hardware
//
#include <hip/hip_runtime.h>
#include <hip/hip_bf16.h>

#define B_SZ    16
#define OUT_LEN 1024
#define IN_LEN  2048
#define DIM     512

typedef __attribute__((ext_vector_type(16))) __bf16 v16bf;
typedef __attribute__((ext_vector_type(8)))  __bf16 v8bf;
typedef __attribute__((ext_vector_type(8)))  float  v8f;
typedef __attribute__((ext_vector_type(4)))  float  v4f;
typedef __attribute__((ext_vector_type(4)))  unsigned int v4u;
typedef __attribute__((ext_vector_type(8)))  int    v8i;
typedef __attribute__((ext_vector_type(4)))  int    v4i;

// LDS layout (manual offsets so TDM descriptors can carry LDS byte addresses)
#define SS_OFF    0                             // float  sS[16][IN_LEN]   : 128 KB
#define SQ_OFF    (16 * IN_LEN * 4)             // __bf16 sQ[16][DIM]      :  16 KB
#define STG0_OFF  (SQ_OFF + 16 * DIM * 2)       // float  stg0[32][DIM]    :  64 KB
#define STG1_OFF  (STG0_OFF + 32 * DIM * 4)     // float  stg1[32][DIM]    :  64 KB
#define LDS_BYTES (STG1_OFF + 32 * DIM * 4)     // 278528 B = 272 KB

static __device__ __forceinline__ __bf16 f2bf(float f) { return (__bf16)f; }

static __device__ __forceinline__ v8f wmma_bf16(v16bf a, v16bf b, v8f c) {
    return __builtin_amdgcn_wmma_f32_16x16x32_bf16(false, a, false, b,
                                                   (short)0, c, false, false);
}
static __device__ __forceinline__ v16bf cat8(v8bf lo, v8bf hi) {
    return __builtin_shufflevector(lo, hi, 0, 1, 2, 3, 4, 5, 6, 7,
                                           8, 9, 10, 11, 12, 13, 14, 15);
}

// TDM: 2D tile load (32 rows x 512 cols of f32, row stride 512) into LDS.
static __device__ __forceinline__ void tdm_load_32x512(const float* gsrc,
                                                       unsigned lds_off) {
    unsigned long long ga = (unsigned long long)(uintptr_t)gsrc;
    // D# group0: count=1 | lds_addr | global_addr[56:0] | type=2  (ISA 8.3)
    v4u g0 = { 1u,
               lds_off,
               (unsigned)(ga & 0xFFFFFFFFu),
               (unsigned)(((ga >> 32) & 0x01FFFFFFu) | (2u << 30)) };
    // D# group1 (ISA 8.4): data_size=2(4B); tensor_dim0=512; tensor_dim1=2048;
    // tile_dim0=512; tile_dim1=32; tile_dim2=0; dim0_stride=512
    v8i g1 = { 0x00020000,          // w0: data_size=2 in bits[17:16]
               0x02000000,          // w1[31:16] = tensor_dim0 = 512
               0x08000000,          // w2[31:16] = tensor_dim1 = 2048
               0x02000000,          // w3[31:16] = tile_dim0 = 512
               32,                  // w4[15:0]  = tile_dim1 = 32
               512,                 // w5        = tensor_dim0_stride = 512
               0, 0 };
    v4i z4 = { 0, 0, 0, 0 };
#if defined(__clang_major__) && (__clang_major__ >= 23)
    v8i z8 = { 0, 0, 0, 0, 0, 0, 0, 0 };
    __builtin_amdgcn_tensor_load_to_lds(g0, g1, z4, z4, z8, 0);
#else
    __builtin_amdgcn_tensor_load_to_lds(g0, g1, z4, z4, 0);
#endif
}

// One workgroup = one (batch, 16-row output tile). 8 waves / 256 threads.
__global__ __launch_bounds__(256) void attn_fused(
    const float* __restrict__ outp,   // [B, OUT_LEN, DIM]
    const float* __restrict__ ctx,    // [B, IN_LEN, DIM]
    const float* __restrict__ W,      // [2*DIM, DIM]
    const float* __restrict__ bias,   // [DIM]
    float* __restrict__ o_out,        // [B, OUT_LEN, DIM]
    float* __restrict__ o_attn)       // [B, OUT_LEN, IN_LEN]
{
    __shared__ __align__(16) unsigned char ldsbuf[LDS_BYTES];
    float  (*sS)[IN_LEN] = (float  (*)[IN_LEN])(ldsbuf + SS_OFF);
    __bf16 (*sQ)[DIM]    = (__bf16 (*)[DIM])  (ldsbuf + SQ_OFF);

    const int tid  = threadIdx.x;
    const int lane = tid & 31;
    const int wave = tid >> 5;
    const int n    = lane & 15;   // col (B/C/D) or row (A) index
    const int hi   = lane >> 4;   // lane-half

    const int batch = blockIdx.x >> 6;          // 64 tiles per batch
    const int o0    = (blockIdx.x & 63) << 4;

    const float* Qg = outp + ((size_t)batch * OUT_LEN + o0) * DIM;
    const float* Cg = ctx  + (size_t)batch * IN_LEN * DIM;

    // ---- Phase 0: load Q tile, hw-convert f32 -> bf16 into LDS ----
    for (int idx = tid; idx < 16 * DIM; idx += 256) {
        int r = idx >> 9, d = idx & (DIM - 1);
        sQ[r][d] = f2bf(Qg[(size_t)r * DIM + d]);
    }
    __syncthreads();

    // ---- Phase 1: S = Q * K^T  (each wave owns 16x16 S tiles) ----
    for (int it = wave; it < IN_LEN / 16; it += 8) {
        const int i0 = it << 4;
        v8f acc = {};
        const float* crow = Cg + (size_t)(i0 + n) * DIM + hi * 16; // B: col n = ctx row
        __builtin_prefetch(crow + (size_t)8 * 16 * DIM, 0, 0);     // next it for this wave
        for (int dc = 0; dc < DIM / 32; ++dc) {
            const int d0 = dc << 5;
            v16bf a = cat8(*(const v8bf*)&sQ[n][d0 + hi * 16],
                           *(const v8bf*)&sQ[n][d0 + hi * 16 + 8]);
            v4f c0 = *(const v4f*)(crow + d0);
            v4f c1 = *(const v4f*)(crow + d0 + 4);
            v4f c2 = *(const v4f*)(crow + d0 + 8);
            v4f c3 = *(const v4f*)(crow + d0 + 12);
            v16bf b;
#pragma unroll
            for (int s = 0; s < 4; ++s) {
                b[s]      = f2bf(c0[s]);
                b[s + 4]  = f2bf(c1[s]);
                b[s + 8]  = f2bf(c2[s]);
                b[s + 12] = f2bf(c3[s]);
            }
            acc = wmma_bf16(a, b, acc);
        }
#pragma unroll
        for (int v = 0; v < 8; ++v)
            sS[v + hi * 8][i0 + n] = acc[v];
    }
    __syncthreads();

    // ---- Phase 2: softmax per row (wave32 reductions), write attn ----
    for (int rr = 0; rr < 2; ++rr) {
        const int r = wave * 2 + rr;
        float mx = -3.4e38f;
        for (int i = lane; i < IN_LEN; i += 32) mx = fmaxf(mx, sS[r][i]);
#pragma unroll
        for (int off = 16; off > 0; off >>= 1) mx = fmaxf(mx, __shfl_xor(mx, off, 32));
        float sum = 0.f;
        for (int i = lane; i < IN_LEN; i += 32) {
            float e = __expf(sS[r][i] - mx);
            sS[r][i] = e;
            sum += e;
        }
#pragma unroll
        for (int off = 16; off > 0; off >>= 1) sum += __shfl_xor(sum, off, 32);
        const float inv = 1.0f / sum;
        float* arow = o_attn + ((size_t)batch * OUT_LEN + o0 + r) * IN_LEN;
        for (int i = lane; i < IN_LEN; i += 32) {
            float p = sS[r][i] * inv;
            sS[r][i] = p;
            arow[i]  = p;
        }
    }
    __syncthreads();

    // ---- Phase 3: mix = P * context, TDM double-buffered 32-row streaming ----
    v8f macc[4] = {};
    if (wave == 0) tdm_load_32x512(Cg, STG0_OFF);                 // prime buffer 0
    for (int ic = 0; ic < IN_LEN / 32; ++ic) {
        const int i0 = ic << 5;
        if (wave == 0) {
            if (ic + 1 < IN_LEN / 32) {
                tdm_load_32x512(Cg + (size_t)(i0 + 32) * DIM,     // issue next tile
                                (ic & 1) ? STG0_OFF : STG1_OFF);
                __builtin_amdgcn_s_wait_tensorcnt(1);             // oldest done
            } else {
                __builtin_amdgcn_s_wait_tensorcnt(0);
            }
        }
        __syncthreads();
        const float (*sC)[DIM] =
            (const float (*)[DIM])(ldsbuf + ((ic & 1) ? STG1_OFF : STG0_OFF));

        v16bf a;                              // P fragment, shared by the wave's 4 d-tiles
#pragma unroll
        for (int s = 0; s < 8; ++s) {
            a[s]     = f2bf(sS[n][i0 + hi * 8 + s]);
            a[s + 8] = f2bf(sS[n][i0 + 16 + hi * 8 + s]);
        }
#pragma unroll
        for (int q = 0; q < 4; ++q) {
            const int d0 = (wave * 4 + q) << 4;
            v16bf b;
#pragma unroll
            for (int s = 0; s < 16; ++s)
                b[s] = f2bf(sC[(s < 8 ? hi * 8 + s : 8 + hi * 8 + s)][d0 + n]);
            macc[q] = wmma_bf16(a, b, macc[q]);
        }
        __syncthreads();
    }

    // Stash mix into (now-dead) sS storage as 16x512 f32.
    float (*sMix)[DIM] = (float (*)[DIM])(ldsbuf + SS_OFF);
#pragma unroll
    for (int q = 0; q < 4; ++q) {
        const int d0 = (wave * 4 + q) << 4;
#pragma unroll
        for (int v = 0; v < 8; ++v)
            sMix[v + hi * 8][d0 + n] = macc[q][v];
    }
    __syncthreads();

    // ---- Phase 4: out = tanh([mix, Q] @ W + b) ----
    // K outer loop; each 32xDIM W k-slice is TDM-staged once and shared by all
    // waves (it was previously re-read strided from global by every wave).
    v8f acc4[4] = {};
    if (wave == 0) tdm_load_32x512(W, STG0_OFF);                  // prime buffer 0
    for (int kc = 0; kc < (2 * DIM) / 32; ++kc) {
        const int k0 = kc << 5;
        if (wave == 0) {
            if (kc + 1 < (2 * DIM) / 32) {
                tdm_load_32x512(W + (size_t)(k0 + 32) * DIM,
                                (kc & 1) ? STG0_OFF : STG1_OFF);
                __builtin_amdgcn_s_wait_tensorcnt(1);
            } else {
                __builtin_amdgcn_s_wait_tensorcnt(0);
            }
        }
        __syncthreads();
        const float (*sW)[DIM] =
            (const float (*)[DIM])(ldsbuf + ((kc & 1) ? STG1_OFF : STG0_OFF));

        v16bf a;                              // combined[:, k0:k0+32] fragment
        if (k0 < DIM) {                       // from mix (f32 in LDS)
#pragma unroll
            for (int s = 0; s < 8; ++s) {
                a[s]     = f2bf(sMix[n][k0 + hi * 8 + s]);
                a[s + 8] = f2bf(sMix[n][k0 + 16 + hi * 8 + s]);
            }
        } else {                              // from Q (bf16 in LDS, vector loads)
            const int kq = k0 - DIM;
            a = cat8(*(const v8bf*)&sQ[n][kq + hi * 8],
                     *(const v8bf*)&sQ[n][kq + 16 + hi * 8]);
        }
#pragma unroll
        for (int q = 0; q < 4; ++q) {
            const int n0 = (wave * 4 + q) << 4;
            v16bf b;
#pragma unroll
            for (int s = 0; s < 16; ++s)
                b[s] = f2bf(sW[(s < 8 ? hi * 8 + s : 8 + hi * 8 + s)][n0 + n]);
            acc4[q] = wmma_bf16(a, b, acc4[q]);
        }
        __syncthreads();
    }

    float* Og = o_out + ((size_t)batch * OUT_LEN + o0) * DIM;
#pragma unroll
    for (int q = 0; q < 4; ++q) {
        const int n0 = (wave * 4 + q) << 4;
        const float bv = bias[n0 + n];
#pragma unroll
        for (int v = 0; v < 8; ++v)
            Og[(size_t)(v + hi * 8) * DIM + n0 + n] = tanhf(acc4[q][v] + bv);
    }
}

extern "C" void kernel_launch(void* const* d_in, const int* in_sizes, int n_in,
                              void* d_out, int out_size, void* d_ws, size_t ws_size,
                              hipStream_t stream) {
    const float* outp = (const float*)d_in[0];
    const float* ctx  = (const float*)d_in[1];
    const float* W    = (const float*)d_in[2];
    const float* bias = (const float*)d_in[3];
    float* o_out  = (float*)d_out;                                    // [16,1024,512]
    float* o_attn = (float*)d_out + (size_t)B_SZ * OUT_LEN * DIM;     // [16,1024,2048]

    dim3 grid(B_SZ * (OUT_LEN / 16));   // 1024 workgroups
    dim3 block(256);                    // 8 wave32
    hipLaunchKernelGGL(attn_fused, grid, block, 0, stream,
                       outp, ctx, W, bias, o_out, o_attn);
}
